// GatedGRU_15985868276229
// MI455X (gfx1250) — compile-verified, matrix operand order checked by reading, not run on hardware
//
#include <hip/hip_runtime.h>
#include <hip/hip_bf16.h>

// ---------------------------------------------------------------------------
// GRU on MI455X (gfx1250, wave32, WMMA).
//   B=128, T=512, D=512, H=1024.
//   Fused formulation: per step t,
//     z   = sigmoid([x_t | h]    @ [W_iu;W_hu] + b_u)
//     rh  = sigmoid([x_t | h]    @ [W_ir;W_hr] + b_r) * h
//     h'  = (1-z)*h + z*tanh([x_t | rh] @ [W_ih;W_hh] + b_h)
//   v_wmma_f32_16x16x32_bf16, f32 accumulate, f32 state.
//   128-thread blocks (4 wave32), 64x64 tiles, double-buffered LDS,
//   K-loop unrolled x2 so buffer parity constant-folds.
// ---------------------------------------------------------------------------

#define Bc 128
#define Tc 512
#define Dc 512
#define Hc 1024
#define KZ (Dc + Hc)      // 1536
#define NK (KZ / 32)      // 48 K-chunks

typedef __attribute__((ext_vector_type(16))) __bf16 v16bf;
typedef __attribute__((ext_vector_type(8)))  __bf16 v8bf;
typedef __attribute__((ext_vector_type(8)))  float  v8f;
typedef __attribute__((ext_vector_type(4)))  float  f4;

union Frag16 { v16bf v; v8bf h[2]; };

// Fast sigmoid / tanh: single v_exp + single v_rcp.
__device__ __forceinline__ float fast_sigmoid(float x) {
    return __builtin_amdgcn_rcpf(1.0f + __expf(-x));
}
__device__ __forceinline__ float fast_tanh(float x) {
    return 1.0f - 2.0f * __builtin_amdgcn_rcpf(__expf(2.0f * x) + 1.0f);
}

// One 16x32 bf16 WMMA operand fragment from a row-major [rows x 32] LDS tile.
// ISA layout (CDNA5 ISA, 16-bit A 16x32):
//   lanes 0-15 : row = lane,    VGPR0-3 = K 0..7,  VGPR4-7 = K 16..23
//   lanes 16-31: row = lane-16, VGPR0-3 = K 8..15, VGPR4-7 = K 24..31
// => exactly two ds_load_b128 per lane.
__device__ __forceinline__ v16bf load_frag(const __bf16* lds, int row, int lane) {
    Frag16 f;
    const int koff = (lane & 16) ? 8 : 0;
    const __bf16* p = lds + row * 32 + koff;
    f.h[0] = *(const v8bf*)(p);
    f.h[1] = *(const v8bf*)(p + 16);
    return f.v;
}

// Pack [W_i (D,H); W_h (H,H)] -> bf16 transposed panel Wt[n*KZ + k].
__global__ __launch_bounds__(256)
void gru_prep(const float* __restrict__ W_iu, const float* __restrict__ W_hu,
              const float* __restrict__ W_ir, const float* __restrict__ W_hr,
              const float* __restrict__ W_ih, const float* __restrict__ W_hh,
              __bf16* __restrict__ Wz, __bf16* __restrict__ Wr,
              __bf16* __restrict__ Wh) {
    size_t idx = (size_t)blockIdx.x * 256 + threadIdx.x;
    int which = (int)(idx / ((size_t)Hc * KZ));
    int rem   = (int)(idx % ((size_t)Hc * KZ));
    int n = rem / KZ, k = rem % KZ;
    const float* Wi  = (which == 0) ? W_iu : (which == 1) ? W_ir : W_ih;
    const float* Whp = (which == 0) ? W_hu : (which == 1) ? W_hr : W_hh;
    __bf16* dst = (which == 0) ? Wz : (which == 1) ? Wr : Wh;
    float v = (k < Dc) ? Wi[(size_t)k * Hc + n] : Whp[(size_t)(k - Dc) * Hc + n];
    dst[(size_t)n * KZ + k] = (__bf16)v;
}

// ---------------------------------------------------------------------------
// Shared GEMM body: C(64 x 64) = [x_t | aux](rows row0..row0+63, K=KZ) @ Wt^T.
// 128 threads = 4 wave32; each wave owns a 16x64 strip (4 v8f accumulators).
// Double-buffered LDS pipeline; K-loop unrolled x2.
//   row0 from blockIdx.y, n0/Wt set by caller, t = time step.
// ---------------------------------------------------------------------------
#define GEMM_BODY(AUX_PTR)                                                      \
    __shared__ __align__(16) __bf16 As[2][64 * 32];                             \
    __shared__ __align__(16) __bf16 Bs[2][64 * 32];                             \
    const int tid  = threadIdx.x;                                               \
    const int lane = tid & 31;                                                  \
    const int wid  = tid >> 5;                                                  \
    const int row0 = blockIdx.y * 64;                                           \
    v8f acc[4] = {v8f{}, v8f{}, v8f{}, v8f{}};                                  \
    const int arow = tid >> 1;       /* 0..63 */                                \
    const int acol = (tid & 1) * 16; /* 0 or 16 */                              \
    f4   areg[4];                                                               \
    v8bf breg[2];                                                               \
    /* stage chunk 0 into buffer 0 */                                           \
    {                                                                           \
        const int k = acol;                                                     \
        const float* srcA = (k < Dc)                                            \
            ? (x + (size_t)(row0 + arow) * Tc * Dc + (size_t)t * Dc + k)        \
            : (AUX_PTR + (size_t)(row0 + arow) * Hc + (k - Dc));                \
        areg[0] = *(const f4*)(srcA + 0);  areg[1] = *(const f4*)(srcA + 4);    \
        areg[2] = *(const f4*)(srcA + 8);  areg[3] = *(const f4*)(srcA + 12);   \
        const __bf16* srcB = Wt + (size_t)(n0 + arow) * KZ + acol;              \
        breg[0] = *(const v8bf*)(srcB);  breg[1] = *(const v8bf*)(srcB + 8);    \
        _Pragma("unroll")                                                       \
        for (int q = 0; q < 4; ++q)                                             \
            _Pragma("unroll")                                                   \
            for (int i = 0; i < 4; ++i)                                         \
                As[0][arow * 32 + acol + q * 4 + i] = (__bf16)areg[q][i];       \
        *(v8bf*)&Bs[0][arow * 32 + acol]     = breg[0];                         \
        *(v8bf*)&Bs[0][arow * 32 + acol + 8] = breg[1];                         \
    }                                                                           \
    __syncthreads();                                                            \
    _Pragma("unroll 2")                                                         \
    for (int kc = 0; kc < NK; ++kc) {                                           \
        const int buf = kc & 1;                                                 \
        if (kc + 1 < NK) { /* prefetch next chunk into registers */             \
            const int kb = (kc + 1) * 32;                                       \
            const int k  = kb + acol;                                           \
            const float* srcA = (k < Dc)                                        \
                ? (x + (size_t)(row0 + arow) * Tc * Dc + (size_t)t * Dc + k)    \
                : (AUX_PTR + (size_t)(row0 + arow) * Hc + (k - Dc));            \
            areg[0] = *(const f4*)(srcA + 0);  areg[1] = *(const f4*)(srcA + 4);\
            areg[2] = *(const f4*)(srcA + 8);  areg[3] = *(const f4*)(srcA + 12);\
            const __bf16* srcB = Wt + (size_t)(n0 + arow) * KZ + kb + acol;     \
            breg[0] = *(const v8bf*)(srcB);  breg[1] = *(const v8bf*)(srcB + 8);\
        }                                                                       \
        const v16bf a = load_frag(&As[buf][0], wid * 16 + (lane & 15), lane);   \
        _Pragma("unroll")                                                       \
        for (int c = 0; c < 4; ++c) {                                           \
            const v16bf b = load_frag(&Bs[buf][0], c * 16 + (lane & 15), lane); \
            acc[c] = __builtin_amdgcn_wmma_f32_16x16x32_bf16(                   \
                false, a, false, b, (short)0, acc[c], false, false);            \
        }                                                                       \
        if (kc + 1 < NK) { /* spill prefetched chunk to the other buffer */     \
            const int nb = (kc + 1) & 1;                                        \
            _Pragma("unroll")                                                   \
            for (int q = 0; q < 4; ++q)                                         \
                _Pragma("unroll")                                               \
                for (int i = 0; i < 4; ++i)                                     \
                    As[nb][arow * 32 + acol + q * 4 + i] = (__bf16)areg[q][i];  \
            *(v8bf*)&Bs[nb][arow * 32 + acol]     = breg[0];                    \
            *(v8bf*)&Bs[nb][arow * 32 + acol + 8] = breg[1];                    \
        }                                                                       \
        __syncthreads();                                                        \
    }

// Phase 1 of step t: z and r*h.  grid = (2*Hc/64, Bc/64), block = 128.
__global__ __launch_bounds__(128)
void gru_gates(const float* __restrict__ x, const float* __restrict__ hbuf,
               const __bf16* __restrict__ Wzp, const __bf16* __restrict__ Wrp,
               const float* __restrict__ b_u, const float* __restrict__ b_r,
               float* __restrict__ zbuf, float* __restrict__ rhbuf, int t) {
    const bool is_r = blockIdx.x >= (Hc / 64);
    const int  n0   = (blockIdx.x - (is_r ? Hc / 64 : 0)) * 64;
    const __bf16* Wt  = is_r ? Wrp : Wzp;
    const float*  bia = is_r ? b_r : b_u;

    GEMM_BODY(hbuf)

    const int m0 = row0 + wid * 16 + ((lane & 16) ? 8 : 0);
    #pragma unroll
    for (int c = 0; c < 4; ++c) {
        const int n = n0 + c * 16 + (lane & 15);
        const float bv = bia[n];
        #pragma unroll
        for (int v = 0; v < 8; ++v) {
            const int m = m0 + v;
            const float s = fast_sigmoid(acc[c][v] + bv);
            if (!is_r) zbuf[(size_t)m * Hc + n] = s;
            else       rhbuf[(size_t)m * Hc + n] = s * hbuf[(size_t)m * Hc + n];
        }
    }
}

// Phase 2 of step t: h' = (1-z)h + z*tanh(...).  grid = (Hc/64, Bc/64).
__global__ __launch_bounds__(128)
void gru_update(const float* __restrict__ x, const float* __restrict__ rhbuf,
                const __bf16* __restrict__ Whp, const float* __restrict__ b_h,
                const float* __restrict__ zbuf, float* __restrict__ hbuf,
                float* __restrict__ out, int t) {
    const int n0 = blockIdx.x * 64;
    const __bf16* Wt = Whp;

    GEMM_BODY(rhbuf)

    const int m0 = row0 + wid * 16 + ((lane & 16) ? 8 : 0);
    #pragma unroll
    for (int c = 0; c < 4; ++c) {
        const int n = n0 + c * 16 + (lane & 15);
        const float bv = b_h[n];
        #pragma unroll
        for (int v = 0; v < 8; ++v) {
            const int m = m0 + v;
            const size_t e = (size_t)m * Hc + n;
            const float ht = fast_tanh(acc[c][v] + bv);
            const float zv = zbuf[e];
            const float hp = hbuf[e];
            const float hn = (1.0f - zv) * hp + zv * ht;
            out[(size_t)t * Bc * Hc + e] = hn;        // hidden_states[t]
            hbuf[e] = hn;
            if (t == Tc - 1)
                out[(size_t)Tc * Bc * Hc + e] = hn;   // h_last
        }
    }
}

extern "C" void kernel_launch(void* const* d_in, const int* in_sizes, int n_in,
                              void* d_out, int out_size, void* d_ws, size_t ws_size,
                              hipStream_t stream) {
    (void)in_sizes; (void)n_in; (void)out_size; (void)ws_size;
    const float* x          = (const float*)d_in[0];
    const float* init_state = (const float*)d_in[1];
    const float* W_iu = (const float*)d_in[2];
    const float* W_hu = (const float*)d_in[3];
    const float* b_u  = (const float*)d_in[4];
    const float* W_ir = (const float*)d_in[5];
    const float* W_hr = (const float*)d_in[6];
    const float* b_r  = (const float*)d_in[7];
    const float* W_ih = (const float*)d_in[8];
    const float* W_hh = (const float*)d_in[9];
    const float* b_h  = (const float*)d_in[10];
    float* out = (float*)d_out;

    // Workspace layout (~11 MB total).
    __bf16* Wz = (__bf16*)d_ws;                 // Hc*KZ bf16
    __bf16* Wr = Wz + (size_t)Hc * KZ;
    __bf16* Wh = Wr + (size_t)Hc * KZ;
    float* hbuf  = (float*)(Wh + (size_t)Hc * KZ);  // Bc*Hc f32
    float* zbuf  = hbuf + (size_t)Bc * Hc;
    float* rhbuf = zbuf + (size_t)Bc * Hc;

    // Pack weights to transposed bf16 panels.
    gru_prep<<<(3u * Hc * KZ) / 256, 256, 0, stream>>>(
        W_iu, W_hu, W_ir, W_hr, W_ih, W_hh, Wz, Wr, Wh);

    // h <- init_state
    hipMemcpyAsync(hbuf, init_state, (size_t)Bc * Hc * sizeof(float),
                   hipMemcpyDeviceToDevice, stream);

    const dim3 blk(128);
    const dim3 grid_g(2 * Hc / 64, Bc / 64);   // 32 x 2 = 64 blocks
    const dim3 grid_u(Hc / 64, Bc / 64);       // 16 x 2 = 32 blocks
    for (int t = 0; t < Tc; ++t) {
        gru_gates<<<grid_g, blk, 0, stream>>>(
            x, hbuf, Wz, Wr, b_u, b_r, zbuf, rhbuf, t);
        gru_update<<<grid_u, blk, 0, stream>>>(
            x, rhbuf, Wh, b_h, zbuf, hbuf, out, t);
    }
}
